// VectorQuantizer2_34634616275541
// MI455X (gfx1250) — compile-verified, matrix operand order checked by reading, not run on hardware
//
#include <hip/hip_runtime.h>
#include <hip/hip_bf16.h>

// CDNA5 / gfx1250 VQ-VAE vector quantizer.
// d = ||z||^2 + ||e||^2 - 2*z.e  via split-bf16 WMMA (hi*hi + lo*hi + hi*lo),
// argmin via cross-lane shfl reduction, fused d/argmin pass, then gather+loss.

typedef __attribute__((ext_vector_type(16))) __bf16 v16bf;
typedef __attribute__((ext_vector_type(8)))  float  v8f;

#define E_DIM 256
#define N_E   1024
#define NTOK  32768     // 32 * 32 * 32 tokens
#define BETA  0.25f

// d_out flat layout (reference tuple order): z_q | loss | min_idx | d
#define ZQ_OFF   ((size_t)0)
#define LOSS_OFF ((size_t)8388608)
#define IDX_OFF  ((size_t)8388609)
#define D_OFF    ((size_t)8421377)

static __device__ __forceinline__ unsigned short bf_bits(__bf16 h) {
    return __builtin_bit_cast(unsigned short, h);
}

// ---------------------------------------------------------------- kernel 0
// enorm[k] = ||embedding[k]||^2  (1024 codes, exact fp32)
__global__ __launch_bounds__(256) void vq_enorm(const float* __restrict__ emb,
                                                float* __restrict__ enorm) {
    int k = blockIdx.x * 256 + threadIdx.x;
    const float* row = emb + (size_t)k * E_DIM;
    float s = 0.f;
    #pragma unroll 8
    for (int c = 0; c < E_DIM; ++c) { float e = row[c]; s += e * e; }
    enorm[k] = s;
}

// ---------------------------------------------------------------- kernel 1
// Per workgroup: 8 waves x 16 tokens = 128 tokens. A (tokens, split bf16)
// lives in VGPRs; each of 64 code tiles (16 codes x 256 feats) is staged once
// into LDS pre-swizzled into the WMMA 16-bit B-fragment layout.
__global__ __launch_bounds__(256) void vq_dist(const float* __restrict__ z,
                                               const float* __restrict__ emb,
                                               float* __restrict__ d_mat,
                                               const float* __restrict__ enorm,
                                               int* __restrict__ idx_ws) {
    __shared__ v16bf lds_bhi[8 * 32];   // [k-chunk][lane] B fragment, hi part
    __shared__ v16bf lds_blo[8 * 32];   // lo part
    unsigned int* bhi32 = (unsigned int*)lds_bhi;
    unsigned int* blo32 = (unsigned int*)lds_blo;

    const int t    = threadIdx.x;
    const int wave = t >> 5;
    const int lane = t & 31;
    const int lo16 = lane & 15;
    const int half = lane >> 4;         // 0: lanes 0-15, 1: lanes 16-31
    const int k8   = half * 8;          // K sub-block per 16-bit A/B layout

    const int tokbase = blockIdx.x * 128 + wave * 16;
    const int token   = tokbase + lo16; // this lane's A row (M = lane%16)
    const int b   = token >> 10;        // H*W = 1024
    const int hwi = token & 1023;
    // z is NCHW: z[b][k][hw], feature stride = 1024 floats
    const float* zb = z + (((size_t)b * E_DIM) << 10) + hwi;

    // ---- load A tile as split bf16 (hi + residual lo), accumulate ||z||^2
    v16bf a_hi[8], a_lo[8];
    float zpart = 0.f;
    #pragma unroll
    for (int c = 0; c < 8; ++c) {
        #pragma unroll
        for (int j = 0; j < 16; ++j) {
            // v16bf element j -> K = k8 + j (j<8) else 16 + k8 + (j-8)
            int k = c * 32 + ((j < 8) ? (k8 + j) : (16 + k8 + (j - 8)));
            float x = zb[(size_t)k << 10];
            zpart += x * x;
            __bf16 h = (__bf16)x;
            __bf16 l = (__bf16)(x - (float)h);
            a_hi[c][j] = h;
            a_lo[c][j] = l;
        }
    }
    // lanes l and l^16 hold complementary K halves of the same token
    float zn_lane = zpart + __shfl_xor(zpart, 16, 32);
    // D-tile: VGPR r belongs to token (r + half*8); fetch that token's norm
    float zn[8];
    #pragma unroll
    for (int r = 0; r < 8; ++r) zn[r] = __shfl(zn_lane, r + half * 8, 32);

    float best[8]; int bidx[8];
    #pragma unroll
    for (int r = 0; r < 8; ++r) { best[r] = 3.402823466e38f; bidx[r] = 0; }

    for (int ct = 0; ct < 64; ++ct) {
        const int code_base = ct * 16;
        __syncthreads();
        // ---- cooperative stage: 16 codes x 256 feats -> LDS B fragments
        {
            const int n  = t >> 4;            // code within tile (B column N)
            const int kr = (t & 15) << 4;     // 16-feature run per thread
            const float* er = emb + (size_t)(code_base + n) * E_DIM + kr;
            #pragma unroll
            for (int p = 0; p < 8; ++p) {
                float f0 = er[p * 2], f1 = er[p * 2 + 1];
                __bf16 h0 = (__bf16)f0, h1 = (__bf16)f1;
                __bf16 l0 = (__bf16)(f0 - (float)h0);
                __bf16 l1 = (__bf16)(f1 - (float)h1);
                int k  = kr + p * 2;
                int c  = k >> 5, kk = k & 31;
                int lb, i;                    // target lane, dword index
                if      (kk <  8) { lb = n;      i = kk >> 1; }
                else if (kk < 16) { lb = n + 16; i = (kk -  8) >> 1; }
                else if (kk < 24) { lb = n;      i = 4 + ((kk - 16) >> 1); }
                else              { lb = n + 16; i = 4 + ((kk - 24) >> 1); }
                int di = (c * 32 + lb) * 8 + i;
                bhi32[di] = (unsigned int)bf_bits(h0) | ((unsigned int)bf_bits(h1) << 16);
                blo32[di] = (unsigned int)bf_bits(l0) | ((unsigned int)bf_bits(l1) << 16);
            }
        }
        __syncthreads();

        // ---- 16x16 cross-term S = z . e  (split-bf16, 3 WMMAs per K-chunk)
        v8f acc = {};
        #pragma unroll
        for (int c = 0; c < 8; ++c) {
            v16bf bh = lds_bhi[c * 32 + lane];
            v16bf bl = lds_blo[c * 32 + lane];
            acc = __builtin_amdgcn_wmma_f32_16x16x32_bf16(
                false, a_hi[c], false, bh, (short)0, acc, false, false);
            acc = __builtin_amdgcn_wmma_f32_16x16x32_bf16(
                false, a_lo[c], false, bh, (short)0, acc, false, false);
            acc = __builtin_amdgcn_wmma_f32_16x16x32_bf16(
                false, a_hi[c], false, bl, (short)0, acc, false, false);
        }

        // ---- d = ||z||^2 + ||e||^2 - 2S ; coalesced store + running argmin
        float en = enorm[code_base + lo16];
        #pragma unroll
        for (int r = 0; r < 8; ++r) {
            float dval = zn[r] + en - 2.0f * acc[r];
            int tok_r = tokbase + r + half * 8;
            d_mat[((size_t)tok_r << 10) + code_base + lo16] = dval;
            if (dval < best[r]) { best[r] = dval; bidx[r] = code_base + lo16; }
        }
    }

    // ---- finalize argmin: min+index across the 16 lanes of each half
    #pragma unroll
    for (int r = 0; r < 8; ++r) {
        float v = best[r]; int i = bidx[r];
        #pragma unroll
        for (int off = 8; off >= 1; off >>= 1) {
            float ov = __shfl_xor(v, off, 32);
            int   oi = __shfl_xor(i, off, 32);
            if (ov < v || (ov == v && oi < i)) { v = ov; i = oi; }
        }
        if (lo16 == 0) idx_ws[tokbase + r + half * 8] = i;
    }
}

// ---------------------------------------------------------------- kernel 2
// Gather z_q (straight-through output == z_q numerically), write min_idx as
// float, accumulate per-block loss partials (deterministic tree reduce).
__global__ __launch_bounds__(256) void vq_gather(const float* __restrict__ z,
                                                 const float* __restrict__ emb,
                                                 const int* __restrict__ idx_ws,
                                                 float* __restrict__ out,
                                                 float* __restrict__ part) {
    __shared__ float red[256];
    const int b = blockIdx.x, q = blockIdx.y, t = threadIdx.x;
    const int hwi   = q * 256 + t;
    const int token = (b << 10) + hwi;
    const int idx   = idx_ws[token];
    out[IDX_OFF + token] = (float)idx;
    const float* er = emb + (size_t)idx * E_DIM;        // L2-resident row
    const float* zp = z + (((size_t)b * E_DIM) << 10) + hwi;
    float* zq = out + ZQ_OFF + (((size_t)b * E_DIM) << 10) + hwi;
    float acc = 0.f;
    #pragma unroll 4
    for (int c = 0; c < E_DIM; ++c) {
        float e  = er[c];
        float zv = zp[(size_t)c << 10];
        zq[(size_t)c << 10] = e;                        // coalesced NCHW write
        float dd = e - zv;
        acc += dd * dd;
    }
    red[t] = acc;
    __syncthreads();
    for (int s = 128; s > 0; s >>= 1) {
        if (t < s) red[t] += red[t + s];
        __syncthreads();
    }
    if (t == 0) part[b * 4 + q] = red[0];
}

// ---------------------------------------------------------------- kernel 3
// loss = (1 + BETA) * mean((z_q - z)^2): both MSE terms are numerically equal.
__global__ __launch_bounds__(128) void vq_loss(const float* __restrict__ part,
                                               float* __restrict__ out) {
    __shared__ float red[128];
    int t = threadIdx.x;
    red[t] = part[t];
    __syncthreads();
    for (int s = 64; s > 0; s >>= 1) {
        if (t < s) red[t] += red[t + s];
        __syncthreads();
    }
    if (t == 0) out[LOSS_OFF] = red[0] * ((1.0f + BETA) / 8388608.0f);
}

// ---------------------------------------------------------------- launcher
extern "C" void kernel_launch(void* const* d_in, const int* in_sizes, int n_in,
                              void* d_out, int out_size, void* d_ws, size_t ws_size,
                              hipStream_t stream) {
    const float* z   = (const float*)d_in[0];   // [32,256,32,32] fp32
    const float* emb = (const float*)d_in[1];   // [1024,256] fp32
    float* out = (float*)d_out;
    float* ws  = (float*)d_ws;
    float* enorm  = ws;                          // 1024 floats
    int*   idx_ws = (int*)(ws + 1024);           // 32768 ints
    float* part   = ws + 1024 + NTOK;            // 128 floats

    vq_enorm<<<dim3(N_E / 256), dim3(256), 0, stream>>>(emb, enorm);
    vq_dist <<<dim3(NTOK / 128), dim3(256), 0, stream>>>(z, emb, out + D_OFF,
                                                         enorm, idx_ws);
    vq_gather<<<dim3(32, 4), dim3(256), 0, stream>>>(z, emb, idx_ws, out, part);
    vq_loss <<<dim3(1), dim3(128), 0, stream>>>(part, out);
}
